// GDNLinearAttention_18528488915154
// MI455X (gfx1250) — matrix operand order repaired and checked by reading, not compiled, not run on hardware
//
#include <hip/hip_runtime.h>

#define HIDDEN      2048
#define NUM_KH      16
#define NUM_VH      32
#define HEAD_DIM    128
#define K_TOTAL     2048   // NUM_KH * HEAD_DIM
#define V_TOTAL     4096   // NUM_VH * HEAD_DIM
#define QKV_TOTAL   8192   // 2*K_TOTAL + V_TOTAL
#define CHUNK       64
#define SCALE_QK    0.08838834764831845f  // 128^-0.5

typedef __attribute__((ext_vector_type(16))) __bf16 v16bf;
typedef __attribute__((ext_vector_type(8)))  __bf16 v8bf;
typedef __attribute__((ext_vector_type(8)))  float  v8f;

static __device__ __forceinline__ unsigned short f2bf_u16(float f) {
  unsigned u = __float_as_uint(f);
  u += 0x7fffu + ((u >> 16) & 1u);   // round to nearest even
  return (unsigned short)(u >> 16);
}

// CDNA5 async memory->LDS DMA (ASYNCcnt-tracked); lds_off = addr[31:0] of the
// generic shared-memory pointer (wave-relative LDS byte offset per ISA 10.2).
static __device__ __forceinline__ void async_load_b128(unsigned lds_off,
                                                       unsigned long long gaddr) {
  asm volatile("global_load_async_to_lds_b128 %0, %1, off"
               :: "v"(lds_off), "v"(gaddr) : "memory");
}
static __device__ __forceinline__ void wait_asynccnt_le8() {
  asm volatile("s_wait_asynccnt 0x8" ::: "memory");
}
static __device__ __forceinline__ void wait_asynccnt_0() {
  asm volatile("s_wait_asynccnt 0x0" ::: "memory");
}

// WMMA fragment loaders (16-bit A 16x32 / B 32x16; contiguous k: lane half h holds k = 16h..16h+15 in A and B)
static __device__ __forceinline__ v16bf load_a_frag(const __bf16* __restrict__ base,
                                                    int row16, int kc, int lda, int lane) {
  const __bf16* p = base + (row16 + (lane & 15)) * lda + kc + ((lane >> 4) << 4);
  v8bf lo = *(const v8bf*)p;
  v8bf hi = *(const v8bf*)(p + 8);
  return __builtin_shufflevector(lo, hi, 0, 1, 2, 3, 4, 5, 6, 7,
                                         8, 9, 10, 11, 12, 13, 14, 15);
}
static __device__ __forceinline__ v16bf load_b_frag(const __bf16* __restrict__ base,
                                                    int col16, int kc, int ldb, int lane) {
  return *(const v16bf*)(base + (col16 + (lane & 15)) * ldb + kc + ((lane >> 4) << 4));
}

// Cooperative LDS GEMM over 8 waves: acc[MTxNT tiles] = A(MxKD)*B(NxKD)^T,
// per-element `store(row, col, val)` callback applies transforms + placement.
template <int MT, int NT, int KD, int LDA, int LDB, typename F>
static __device__ __forceinline__ void lds_gemm(const __bf16* __restrict__ Al,
                                                const __bf16* __restrict__ Bl,
                                                int wid, int lane, F&& store) {
  const v8f vz = {0.f, 0.f, 0.f, 0.f, 0.f, 0.f, 0.f, 0.f};
  for (int tix = wid; tix < MT * NT; tix += 8) {
    const int tm = (tix / NT) * 16;
    const int tn = (tix % NT) * 16;
    v8f acc = vz;
#pragma unroll
    for (int kc = 0; kc < KD; kc += 32) {
      v16bf a = load_a_frag(Al, tm, kc, LDA, lane);
      v16bf b = load_b_frag(Bl, tn, kc, LDB, lane);
      acc = __builtin_amdgcn_wmma_f32_16x16x32_bf16(false, a, false, b, (short)0, acc,
                                                    false, false);
    }
    const int r0 = tm + ((lane >> 4) << 3);
    const int cc = tn + (lane & 15);
#pragma unroll
    for (int r = 0; r < 8; r++) store(r0 + r, cc, acc[r]);
  }
}

// ---------------------------------------------------------------- f32 -> bf16
__global__ __launch_bounds__(256) void k_cvt_bf16(
    const float* __restrict__ in, unsigned short* __restrict__ out, int n) {
  int i = blockIdx.x * blockDim.x + threadIdx.x;
  int stride = gridDim.x * blockDim.x;
  for (; i < n; i += stride) out[i] = f2bf_u16(in[i]);
}

// ------------------------------------------------- WMMA GEMM: C = A(MxK) * B(NxK)^T
// Double-buffered GLOBAL_LOAD_ASYNC_TO_LDS pipeline.
__global__ __launch_bounds__(256) void k_gemm_bf16(
    float* __restrict__ C, const __bf16* __restrict__ A, const __bf16* __restrict__ B,
    int M, int N, int K) {
  constexpr int BM = 128, BN = 128, BK = 64;
  __shared__ __attribute__((aligned(32))) __bf16 ldsA[2][BM * BK];
  __shared__ __attribute__((aligned(32))) __bf16 ldsB[2][BN * BK];

  const int tid  = threadIdx.x;
  const int lane = tid & 31;
  const int wid  = tid >> 5;
  const int wm   = wid >> 1;
  const int wn   = wid & 1;
  const int mBase = blockIdx.y * BM;
  const int nBase = blockIdx.x * BN;

  auto issue_tile = [&](int k0, int b) {
#pragma unroll
    for (int i = 0; i < 4; i++) {
      int c   = tid + i * 256;
      int row = c >> 3;
      int col = (c & 7) << 3;
      async_load_b128((unsigned)(unsigned long long)&ldsA[b][row * BK + col],
                      (unsigned long long)&A[(size_t)(mBase + row) * K + k0 + col]);
      async_load_b128((unsigned)(unsigned long long)&ldsB[b][row * BK + col],
                      (unsigned long long)&B[(size_t)(nBase + row) * K + k0 + col]);
    }
  };

  const v8f vzero = {0.f, 0.f, 0.f, 0.f, 0.f, 0.f, 0.f, 0.f};
  v8f acc[2][4];
#pragma unroll
  for (int i = 0; i < 2; i++)
#pragma unroll
    for (int j = 0; j < 4; j++) acc[i][j] = vzero;

  const int nIter = K / BK;
  issue_tile(0, 0);

  for (int it = 0; it < nIter; ++it) {
    const int b = it & 1;
    __syncthreads();
    if (it + 1 < nIter) {
      issue_tile((it + 1) * BK, b ^ 1);
      wait_asynccnt_le8();
    } else {
      wait_asynccnt_0();
    }
    __syncthreads();

#pragma unroll
    for (int kc = 0; kc < BK; kc += 32) {
      v16bf af[2];
#pragma unroll
      for (int mt = 0; mt < 2; mt++)
        af[mt] = load_a_frag(&ldsA[b][0], wm * 32 + mt * 16, kc, BK, lane);
      v16bf bfr[4];
#pragma unroll
      for (int nt = 0; nt < 4; nt++)
        bfr[nt] = load_b_frag(&ldsB[b][0], wn * 64 + nt * 16, kc, BK, lane);
#pragma unroll
      for (int mt = 0; mt < 2; mt++)
#pragma unroll
        for (int nt = 0; nt < 4; nt++)
          acc[mt][nt] = __builtin_amdgcn_wmma_f32_16x16x32_bf16(
              false, af[mt], false, bfr[nt], (short)0, acc[mt][nt], false, false);
    }
  }

#pragma unroll
  for (int mt = 0; mt < 2; mt++) {
    int row0 = mBase + wm * 32 + mt * 16 + ((lane >> 4) << 3);
#pragma unroll
    for (int nt = 0; nt < 4; nt++) {
      int col = nBase + wn * 64 + nt * 16 + (lane & 15);
      v8f a = acc[mt][nt];
#pragma unroll
      for (int r = 0; r < 8; r++)
        C[(size_t)(row0 + r) * N + col] = a[r];
    }
  }
}

// ------------------------------------- a/b projections + fused GDN gating (tiny GEMM)
__global__ __launch_bounds__(64) void k_proj_ab_gate(
    const float* __restrict__ hidden, const float* __restrict__ Wa, const float* __restrict__ Wb,
    const float* __restrict__ A_log, const float* __restrict__ dt_bias,
    float* __restrict__ g, float* __restrict__ beta) {
  int t = blockIdx.x;
  int j = threadIdx.x;
  const float* hrow = hidden + (size_t)t * HIDDEN;
  const float* w = (j < 32) ? (Wa + (size_t)j * HIDDEN) : (Wb + (size_t)(j - 32) * HIDDEN);
  float a0 = 0.f, a1 = 0.f, a2 = 0.f, a3 = 0.f;
  for (int i = 0; i < HIDDEN; i += 4) {
    float4 hv = *(const float4*)&hrow[i];
    float4 wv = *(const float4*)&w[i];
    a0 = fmaf(hv.x, wv.x, a0);
    a1 = fmaf(hv.y, wv.y, a1);
    a2 = fmaf(hv.z, wv.z, a2);
    a3 = fmaf(hv.w, wv.w, a3);
  }
  float acc = (a0 + a1) + (a2 + a3);
  if (j < 32) {
    float x  = acc + dt_bias[j];
    float sp = (x > 20.f) ? x : log1pf(__expf(x));
    g[(size_t)t * NUM_VH + j] = -__expf(A_log[j]) * sp;
  } else {
    beta[(size_t)t * NUM_VH + (j - 32)] = 1.f / (1.f + __expf(-acc));
  }
}

// ------------------------ causal depthwise conv1d + SiLU + (q,k) L2-norm, v passthrough
__global__ __launch_bounds__(128) void k_conv_silu_norm(
    const float* __restrict__ mixed, const float* __restrict__ convw,
    float* __restrict__ qn, float* __restrict__ kn, float* __restrict__ v, int T) {
  int t  = blockIdx.x;
  int hs = blockIdx.y;
  int j  = threadIdx.x;
  int c0;
  if (hs < 16)      c0 = hs * HEAD_DIM;
  else if (hs < 32) c0 = K_TOTAL + (hs - 16) * HEAD_DIM;
  else              c0 = 2 * K_TOTAL + (hs - 32) * HEAD_DIM;
  int c = c0 + j;
  float4 w = *(const float4*)&convw[c * 4];
  float wv[4] = {w.x, w.y, w.z, w.w};
  float y = 0.f;
#pragma unroll
  for (int jj = 0; jj < 4; jj++) {
    int ts = t - 3 + jj;
    if (ts >= 0) y = fmaf(mixed[(size_t)ts * QKV_TOTAL + c], wv[jj], y);
  }
  y = y / (1.f + __expf(-y));  // SiLU
  if (hs < 32) {
    __shared__ float sb[128];
    sb[j] = y * y;
    __syncthreads();
    for (int s = 64; s > 0; s >>= 1) { if (j < s) sb[j] += sb[j + s]; __syncthreads(); }
    float out = y * rsqrtf(sb[0] + 1e-6f);
    if (hs < 16) qn[(size_t)t * K_TOTAL + hs * HEAD_DIM + j] = out;
    else         kn[(size_t)t * K_TOTAL + (hs - 16) * HEAD_DIM + j] = out;
  } else {
    v[(size_t)t * V_TOTAL + (hs - 32) * HEAD_DIM + j] = y;
  }
}

// --------------------------------------------------------------------------------
// Chunked gated delta rule on WMMA. One block per (v-head, Dv-half); 256 threads.
// State kept transposed in LDS: ST[v][d] (f32) so it serves as a WMMA B operand.
// Per chunk (C=64):  M = tril(beta_s e^{Gs-Gr} K K^T), U = (I+M)^{-1} beta(V - Lam K S0)
//   O = [scale e^{Gt-Gs} qk]_{s<=t} U + scale Lam_t Q S0,  S = e^{GC} S0 + K'^T U.
// All exp arguments are differences of a decreasing cumsum -> <= 0, no overflow.
// --------------------------------------------------------------------------------
__global__ __launch_bounds__(256) void k_gdn_chunked(
    const float* __restrict__ qn, const float* __restrict__ kn, const float* __restrict__ vglob,
    const float* __restrict__ g, const float* __restrict__ beta,
    float* __restrict__ o, int T) {
  const int h    = blockIdx.x >> 1;    // v head
  const int vh   = blockIdx.x & 1;     // Dv half (64 columns)
  const int hk   = h >> 1;             // GQA ratio 2
  const int tid  = threadIdx.x;
  const int lane = tid & 31;
  const int wid  = tid >> 5;
  const size_t obase = (size_t)h * HEAD_DIM + vh * 64;

  __shared__ __attribute__((aligned(32))) float   ST[64 * 128];   // state^T [v][d]  32KB
  __shared__ __attribute__((aligned(32))) __bf16  STb[64 * 128];  // bf16 copy      16KB
  __shared__ __attribute__((aligned(32))) __bf16  Kb[64 * 128];   // K [s][d]       16KB
  __shared__ __attribute__((aligned(32))) __bf16  Qb[64 * 128];   // Q [t][d]       16KB
  __shared__ __attribute__((aligned(32))) __bf16  Ktb[128 * 64];  // K'^T [d][s]    16KB
  __shared__ __attribute__((aligned(32))) float   Mm[64 * 64];    // M              16KB
  __shared__ __attribute__((aligned(32))) __bf16  Pb[64 * 64];    // P bf16          8KB
  __shared__ __attribute__((aligned(32))) float   Rt[64 * 64];    // R^T -> U^T     16KB
  __shared__ __attribute__((aligned(32))) __bf16  Utb[64 * 64];   // U^T bf16        8KB
  __shared__ float Gc[CHUNK];   // inclusive cumsum of g over the chunk
  __shared__ float Bt[CHUNK];   // beta over the chunk

  // zero state
#pragma unroll
  for (int i = 0; i < 32; i++) ST[tid + i * 256] = 0.f;
  __syncthreads();

  const int nChunks = T / CHUNK;
  for (int c = 0; c < nChunks; ++c) {
    const int t0 = c * CHUNK;

    // ---- stage K/Q chunk (bf16), gate cumsum, beta ----
#pragma unroll
    for (int i = 0; i < 32; i++) {
      int e = tid + i * 256;                     // 8192 elements
      int s = e >> 7, d = e & 127;
      ((unsigned short*)Kb)[e] = f2bf_u16(kn[(size_t)(t0 + s) * K_TOTAL + hk * HEAD_DIM + d]);
      ((unsigned short*)Qb)[e] = f2bf_u16(qn[(size_t)(t0 + s) * K_TOTAL + hk * HEAD_DIM + d]);
    }
    if (tid == 0) {
      float run = 0.f;
      for (int s = 0; s < CHUNK; s++) { run += g[(size_t)(t0 + s) * NUM_VH + h]; Gc[s] = run; }
    }
    if (tid >= 64 && tid < 128) Bt[tid - 64] = beta[(size_t)(t0 + tid - 64) * NUM_VH + h];
    __syncthreads();

    // ---- STb = bf16(ST) ; Ktb[d][s] = e^{GC-Gs} k[s][d] ----
    const float GC = Gc[CHUNK - 1];
#pragma unroll
    for (int i = 0; i < 32; i++) {
      int e = tid + i * 256;
      ((unsigned short*)STb)[e] = f2bf_u16(ST[e]);
      int d = e >> 6, s = e & 63;
      float kv = kn[(size_t)(t0 + s) * K_TOTAL + hk * HEAD_DIM + d];
      ((unsigned short*)Ktb)[e] = f2bf_u16(__expf(GC - Gc[s]) * kv);
    }
    __syncthreads();

    // ---- GEMM batch 1: M, P, R (all K-dim 128) ----
    lds_gemm<4, 4, 128, 128, 128>(Kb, Kb, wid, lane, [&](int s, int r, float val) {
      Mm[s * 64 + r] = (r < s) ? Bt[s] * __expf(Gc[s] - Gc[r]) * val : 0.f;
    });
    lds_gemm<4, 4, 128, 128, 128>(Qb, Kb, wid, lane, [&](int t, int s, float val) {
      float pv = (s <= t) ? SCALE_QK * __expf(Gc[t] - Gc[s]) * val : 0.f;
      ((unsigned short*)Pb)[t * 64 + s] = f2bf_u16(pv);
    });
    lds_gemm<4, 4, 128, 128, 128>(Kb, STb, wid, lane, [&](int s, int vc, float val) {
      float vg = vglob[(size_t)(t0 + s) * V_TOTAL + obase + vc];
      Rt[vc * 64 + s] = Bt[s] * (vg - __expf(Gc[s]) * val);
    });
    __syncthreads();

    // ---- forward substitution: U = (I+M)^{-1} R, in place in Rt; thread = v column ----
    if (tid < 64) {
      float* __restrict__ urow = &Rt[tid * 64];
#pragma unroll 1
      for (int gb = 0; gb < 8; gb++) {
        const int s0 = gb * 8;
        float nu[8], acc[8];
#pragma unroll
        for (int i = 0; i < 8; i++) { nu[i] = urow[s0 + i]; acc[i] = 0.f; }
        for (int r = 0; r < s0; r++) {            // dense rectangle, 8-way ILP
          float ur = urow[r];
#pragma unroll
          for (int i = 0; i < 8; i++) acc[i] = fmaf(Mm[(s0 + i) * 64 + r], ur, acc[i]);
        }
#pragma unroll
        for (int i = 0; i < 8; i++) {             // 8x8 triangle
          float ui = nu[i] - acc[i];
#pragma unroll
          for (int p = i + 1; p < 8; p++) acc[p] = fmaf(Mm[(s0 + p) * 64 + (s0 + i)], ui, acc[p]);
          nu[i] = ui;
        }
#pragma unroll
        for (int i = 0; i < 8; i++) urow[s0 + i] = nu[i];
      }
    }
    __syncthreads();

    // ---- Utb = bf16(U^T) ----
#pragma unroll
    for (int i = 0; i < 16; i++) {
      int e = tid + i * 256;                      // 4096 elements
      ((unsigned short*)Utb)[e] = f2bf_u16(Rt[e]);
    }
    __syncthreads();

    // ---- O = P*U + scale*Lam_t*(Q*S0) ; store to global ----
    {
      const v8f vz = {0.f, 0.f, 0.f, 0.f, 0.f, 0.f, 0.f, 0.f};
      for (int tix = wid; tix < 16; tix += 8) {
        const int tm = (tix / 4) * 16, tn = (tix % 4) * 16;
        v8f aI = vz, aE = vz;
#pragma unroll
        for (int kc = 0; kc < 64; kc += 32)
          aI = __builtin_amdgcn_wmma_f32_16x16x32_bf16(
              false, load_a_frag(Pb, tm, kc, 64, lane),
              false, load_b_frag(Utb, tn, kc, 64, lane), (short)0, aI, false, false);
#pragma unroll
        for (int kc = 0; kc < 128; kc += 32)
          aE = __builtin_amdgcn_wmma_f32_16x16x32_bf16(
              false, load_a_frag(Qb, tm, kc, 128, lane),
              false, load_b_frag(STb, tn, kc, 128, lane), (short)0, aE, false, false);
        const int r0 = tm + ((lane >> 4) << 3);
        const int cc = tn + (lane & 15);
#pragma unroll
        for (int r = 0; r < 8; r++) {
          int t = r0 + r;
          float lam = __expf(Gc[t]);
          o[(size_t)(t0 + t) * V_TOTAL + obase + cc] = aI[r] + SCALE_QK * lam * aE[r];
        }
      }
      // ---- S = e^{GC} S0 + K'^T U  (in transposed layout: ST[v][d]) ----
      const float lamC = __expf(GC);
      for (int tix = wid; tix < 32; tix += 8) {
        const int tm = (tix / 8) * 16, tn = (tix % 8) * 16;   // tm: v, tn: d
        v8f a = vz;
#pragma unroll
        for (int kc = 0; kc < 64; kc += 32)
          a = __builtin_amdgcn_wmma_f32_16x16x32_bf16(
              false, load_a_frag(Utb, tm, kc, 64, lane),
              false, load_b_frag(Ktb, tn, kc, 64, lane), (short)0, a, false, false);
        const int r0 = tm + ((lane >> 4) << 3);
        const int cc = tn + (lane & 15);
#pragma unroll
        for (int r = 0; r < 8; r++) {
          int idx = (r0 + r) * 128 + cc;
          ST[idx] = lamC * ST[idx] + a[r];
        }
      }
    }
    __syncthreads();   // chunk done before staging buffers are overwritten
  }
}

// ------------------------ gated RMSNorm * silu(z), emit bf16 for the out-projection
__global__ __launch_bounds__(128) void k_gate_norm(
    const float* __restrict__ o, const float* __restrict__ z, const float* __restrict__ nw,
    unsigned short* __restrict__ ogBF, int T) {
  int t = blockIdx.x, h = blockIdx.y, j = threadIdx.x;
  size_t idx = (size_t)t * V_TOTAL + h * HEAD_DIM + j;
  float ov = o[idx];
  __shared__ float sb[128];
  sb[j] = ov * ov;
  __syncthreads();
  for (int s = 64; s > 0; s >>= 1) { if (j < s) sb[j] += sb[j + s]; __syncthreads(); }
  float inv = rsqrtf(sb[0] * (1.f / 128.f) + 1e-6f);
  float zz  = z[idx];
  float val = ov * inv * nw[j] * (zz / (1.f + __expf(-zz)));
  ogBF[idx] = f2bf_u16(val);
}

// -----------------------------------------------------------------------------
extern "C" void kernel_launch(void* const* d_in, const int* in_sizes, int n_in,
                              void* d_out, int out_size, void* d_ws, size_t ws_size,
                              hipStream_t stream) {
  const float* hidden = (const float*)d_in[0];
  const float* Wqkv   = (const float*)d_in[1];
  const float* Wz     = (const float*)d_in[2];
  const float* Wa     = (const float*)d_in[3];
  const float* Wb     = (const float*)d_in[4];
  const float* convw  = (const float*)d_in[5];
  const float* Wout   = (const float*)d_in[6];
  const float* normw  = (const float*)d_in[7];
  const float* A_log  = (const float*)d_in[8];
  const float* dtb    = (const float*)d_in[9];
  float* out = (float*)d_out;
  const int T = in_sizes[0] / HIDDEN;  // 2048

  char* ws = (char*)d_ws;
  size_t o0 = 0;                                        // hidden bf16
  size_t o1 = o0 + (size_t)T * HIDDEN * 2;              // Wqkv bf16 (later reused as z f32)
  size_t o2 = o1 + (size_t)QKV_TOTAL * HIDDEN * 2;      // Wz bf16
  size_t o3 = o2 + (size_t)V_TOTAL * HIDDEN * 2;        // Wout bf16
  size_t o4 = o3 + (size_t)HIDDEN * V_TOTAL * 2;        // mixed f32 (later o f32 + ogated bf16)
  size_t o5 = o4 + (size_t)T * QKV_TOTAL * 4;           // qn
  size_t o6 = o5 + (size_t)T * K_TOTAL * 4;             // kn
  size_t o7 = o6 + (size_t)T * K_TOTAL * 4;             // v
  size_t o8 = o7 + (size_t)T * V_TOTAL * 4;             // g
  size_t o9 = o8 + (size_t)T * NUM_VH * 4;              // beta

  unsigned short* hidBF  = (unsigned short*)(ws + o0);
  unsigned short* WqkvBF = (unsigned short*)(ws + o1);
  unsigned short* WzBF   = (unsigned short*)(ws + o2);
  unsigned short* WoutBF = (unsigned short*)(ws + o3);
  float* mixed = (float*)(ws + o4);
  float* zbuf  = (float*)(ws + o1);                     // aliases WqkvBF (dead after GEMM-qkv)
  float* qn    = (float*)(ws + o5);
  float* kn    = (float*)(ws + o6);
  float* vbuf  = (float*)(ws + o7);
  float* gbuf  = (float*)(ws + o8);
  float* bbuf  = (float*)(ws + o9);
  float* obuf  = (float*)(ws + o4);                     // aliases mixed (dead after conv)
  unsigned short* ogBF = (unsigned short*)(ws + o4 + (size_t)T * V_TOTAL * 4);

  // 1) bf16 conversions
  k_cvt_bf16<<<4096, 256, 0, stream>>>(hidden, hidBF, T * HIDDEN);
  k_cvt_bf16<<<4096, 256, 0, stream>>>(Wqkv, WqkvBF, QKV_TOTAL * HIDDEN);
  k_cvt_bf16<<<4096, 256, 0, stream>>>(Wz, WzBF, V_TOTAL * HIDDEN);
  k_cvt_bf16<<<4096, 256, 0, stream>>>(Wout, WoutBF, HIDDEN * V_TOTAL);

  // 2) WMMA GEMMs: mixed = H·Wqkvᵀ ; z = H·Wzᵀ
  k_gemm_bf16<<<dim3(QKV_TOTAL / 128, T / 128), 256, 0, stream>>>(
      mixed, (const __bf16*)hidBF, (const __bf16*)WqkvBF, T, QKV_TOTAL, HIDDEN);
  k_gemm_bf16<<<dim3(V_TOTAL / 128, T / 128), 256, 0, stream>>>(
      zbuf, (const __bf16*)hidBF, (const __bf16*)WzBF, T, V_TOTAL, HIDDEN);

  // 3) a/b projections + gating; conv+SiLU+L2-norm
  k_proj_ab_gate<<<T, 64, 0, stream>>>(hidden, Wa, Wb, A_log, dtb, gbuf, bbuf);
  k_conv_silu_norm<<<dim3(T, 64), 128, 0, stream>>>(mixed, convw, qn, kn, vbuf, T);

  // 4) chunked gated delta rule on WMMA: 64 blocks = (head, Dv-half)
  k_gdn_chunked<<<NUM_VH * 2, 256, 0, stream>>>(qn, kn, vbuf, gbuf, bbuf, obuf, T);

  // 5) gated RMSNorm * silu(z) -> bf16 ; out-projection on WMMA
  k_gate_norm<<<dim3(T, NUM_VH), 128, 0, stream>>>(obuf, zbuf, normw, ogBF, T);
  k_gemm_bf16<<<dim3(HIDDEN / 128, T / 128), 256, 0, stream>>>(
      out, (const __bf16*)ogBF, (const __bf16*)WoutBF, T, HIDDEN, V_TOTAL);
}
